// LLMoUAttention_266287973128
// MI455X (gfx1250) — matrix operand order repaired and checked by reading, not compiled, not run on hardware
//
#include <hip/hip_runtime.h>
#include <hip/hip_bf16.h>

#define BATCH 2
#define SEQ   2048
#define HID   2048
#define NHEAD 16
#define HEADD 128
#define ALPHA 0.088388347648318447f   // 1/sqrt(128)

typedef __attribute__((ext_vector_type(16))) __bf16 v16bf;
typedef __attribute__((ext_vector_type(8)))  float  v8f;

union FragAB { v16bf v; uint4 q[2]; };

__device__ __forceinline__ unsigned short f2bf(float f) {
  unsigned int u = __float_as_uint(f);
  u += 0x7FFFu + ((u >> 16) & 1u);          // round-to-nearest-even
  return (unsigned short)(u >> 16);
}

// CDNA5 async global->LDS copy, 16B per lane, tracked by ASYNCcnt.
// ldsAddr: per-lane LDS byte offset (low 32 bits of generic shared pointer).
__device__ __forceinline__ void async_ld16(const void* lds, const void* gptr) {
  unsigned l = (unsigned)(unsigned long long)lds;
  asm volatile("global_load_async_to_lds_b128 %0, %1, off"
               :: "v"(l), "v"(gptr) : "memory");
}
__device__ __forceinline__ void async_wait0() {
  asm volatile("s_wait_asynccnt 0x0" ::: "memory");
}

// ---------------------------------------------------------------- fp32 -> bf16
__global__ void __launch_bounds__(256) cvt_bf16_kernel(const float* __restrict__ src,
                                                       unsigned short* __restrict__ dst,
                                                       int n4) {
  int i = blockIdx.x * 256 + threadIdx.x;
  if (i < n4) {
    float4 f = ((const float4*)src)[i];
    ushort4 o;
    o.x = f2bf(f.x); o.y = f2bf(f.y); o.z = f2bf(f.z); o.w = f2bf(f.w);
    ((ushort4*)dst)[i] = o;
  }
}

// ---------------------------------------------------------------- QKV GEMM
// C[4096 x 6144] = A[4096 x 2048] * B[2048 x 6144] + bias; scatter to k/q/v
__global__ void __launch_bounds__(256) gemm_qkv_kernel(
    const unsigned short* __restrict__ A,
    const unsigned short* __restrict__ Bm,
    const float* __restrict__ bias,
    unsigned short* __restrict__ kb,
    unsigned short* __restrict__ qbuf,
    unsigned short* __restrict__ vb) {
  const int K = HID, N = 3 * HID;
  __shared__ unsigned short As[128][40];   // [m][k], stride 80B (16B aligned)
  __shared__ unsigned short Bs[128][40];   // transposed: [n][k]
  const int tid  = threadIdx.x;
  const int wave = tid >> 5, lane = tid & 31;
  const int half = lane >> 4, l16 = lane & 15;
  const int nBlock = blockIdx.x * 128;
  const int mBlock = blockIdx.y * 128;
  const int wm = (wave & 3) * 32, wn = (wave >> 2) * 64;
  v8f acc[2][4] = {};
  for (int k0 = 0; k0 < K; k0 += 32) {
#pragma unroll
    for (int it = 0; it < 2; ++it) {        // A tile 128x32: async direct to LDS
      int idx = (tid + it * 256) * 8;
      int r = idx >> 5, c = idx & 31;
      async_ld16(&As[r][c], A + (size_t)(mBlock + r) * K + k0 + c);
    }
#pragma unroll
    for (int it = 0; it < 2; ++it) {        // B tile 32x128, stored transposed
      int idx = (tid + it * 256) * 8;
      int kk = idx >> 7, nn = idx & 127;
      uint4 d = *(const uint4*)(Bm + (size_t)(k0 + kk) * N + nBlock + nn);
      const unsigned short* ds = (const unsigned short*)&d;
#pragma unroll
      for (int j = 0; j < 8; ++j) Bs[nn + j][kk] = ds[j];
    }
    async_wait0();
    __syncthreads();
    if (k0 + 32 < K) {                      // stream next K panel into caches
      int idx = tid * 8;
      __builtin_prefetch(A + (size_t)(mBlock + (idx >> 5)) * K + (k0 + 32) + (idx & 31), 0, 1);
      __builtin_prefetch(Bm + (size_t)(k0 + 32 + (idx >> 7)) * N + nBlock + (idx & 127), 0, 1);
    }
    FragAB af[2], bf[4];
#pragma unroll
    for (int mi = 0; mi < 2; ++mi) {
      int r = wm + mi * 16 + l16;
      af[mi].q[0] = *(const uint4*)&As[r][half * 8];
      af[mi].q[1] = *(const uint4*)&As[r][16 + half * 8];
    }
#pragma unroll
    for (int ni = 0; ni < 4; ++ni) {
      int n = wn + ni * 16 + l16;
      bf[ni].q[0] = *(const uint4*)&Bs[n][half * 8];
      bf[ni].q[1] = *(const uint4*)&Bs[n][16 + half * 8];
    }
#pragma unroll
    for (int mi = 0; mi < 2; ++mi)
#pragma unroll
      for (int ni = 0; ni < 4; ++ni)
        acc[mi][ni] = __builtin_amdgcn_wmma_f32_16x16x32_bf16(
            false, af[mi].v, false, bf[ni].v, (short)0, acc[mi][ni], false, false);
    __syncthreads();
  }
  // epilogue: +bias, split col -> (head, which{k,q,v}, d), per-head layout
#pragma unroll
  for (int mi = 0; mi < 2; ++mi) {
#pragma unroll
    for (int ni = 0; ni < 4; ++ni) {
      int col  = nBlock + wn + ni * 16 + l16;
      int head = col / 384;
      int rem  = col - head * 384;
      int which = rem >> 7;                 // 0=k, 1=q, 2=v
      int d     = rem & 127;
      float bia = bias[col];
      unsigned short* dstbuf = (which == 0) ? kb : (which == 1 ? qbuf : vb);
#pragma unroll
      for (int i = 0; i < 8; ++i) {
        int rowg = mBlock + wm + mi * 16 + half * 8 + i;
        int bb = rowg >> 11, ss = rowg & (SEQ - 1);
        size_t di = (((size_t)bb * NHEAD + head) * SEQ + ss) * HEADD + d;
        dstbuf[di] = f2bf(acc[mi][ni][i] + bia);
      }
    }
  }
}

// ---------------------------------------------------------------- flash attention
__global__ void __launch_bounds__(256) attn_kernel(
    const unsigned short* __restrict__ qbuf,
    const unsigned short* __restrict__ kb,
    const unsigned short* __restrict__ vb,
    const float* __restrict__ alibi,
    unsigned short* __restrict__ ctxb) {
  __shared__ unsigned short Ks[64][HEADD + 8];       // [k-row][d]
  __shared__ unsigned short Vt[HEADD][64 + 8];       // transposed: [d][k-row]
  __shared__ unsigned short Ps[8][16][64 + 8];       // per-wave P staging
  const int qt = blockIdx.x, h = blockIdx.y, b = blockIdx.z;
  const int tid  = threadIdx.x;
  const int wave = tid >> 5, lane = tid & 31;
  const int half = lane >> 4, l16 = lane & 15;
  const size_t headOff = ((size_t)b * NHEAD + h) * SEQ * HEADD;
  const unsigned short* qh = qbuf + headOff + (size_t)(qt * 128) * HEADD;
  const unsigned short* kh = kb + headOff;
  const unsigned short* vh = vb + headOff;

  // hoist this wave's 16 Q rows (4 K-chunks of 32) into registers
  FragAB aq[4];
  {
    const unsigned short* qrow = qh + (size_t)(wave * 16 + l16) * HEADD;
#pragma unroll
    for (int kk = 0; kk < 4; ++kk) {
      aq[kk].q[0] = *(const uint4*)(qrow + kk * 32 + half * 8);
      aq[kk].q[1] = *(const uint4*)(qrow + kk * 32 + 16 + half * 8);
    }
  }
  v8f cacc[8] = {};
  float mrun[8], lrun[8];
#pragma unroll
  for (int i = 0; i < 8; ++i) { mrun[i] = -3.0e38f; lrun[i] = 0.0f; }
  const int rowTop = qt * 128 + wave * 16 + 15;
  const int nkt = 2 * qt + 2;                         // causal tile count
  for (int kt = 0; kt < nkt; ++kt) {
#pragma unroll
    for (int it = 0; it < 4; ++it) {                  // K async to LDS, V transposed
      int idx = (tid + it * 256) * 8;
      int r = idx >> 7, c = idx & 127;
      async_ld16(&Ks[r][c], kh + (size_t)(kt * 64 + r) * HEADD + c);
      uint4 dv = *(const uint4*)(vh + (size_t)(kt * 64 + r) * HEADD + c);
      const unsigned short* dsv = (const unsigned short*)&dv;
#pragma unroll
      for (int j = 0; j < 8; ++j) Vt[c + j][r] = dsv[j];
    }
    async_wait0();
    __syncthreads();
    const int colBase = kt * 64;
    if (colBase <= rowTop) {
      // S = Q @ K^T : 4 n-tiles x 4 K-chunks
      v8f sacc[4] = {};
#pragma unroll
      for (int kk = 0; kk < 4; ++kk) {
#pragma unroll
        for (int nt = 0; nt < 4; ++nt) {
          FragAB bk;
          bk.q[0] = *(const uint4*)&Ks[nt * 16 + l16][kk * 32 + half * 8];
          bk.q[1] = *(const uint4*)&Ks[nt * 16 + l16][kk * 32 + 16 + half * 8];
          sacc[nt] = __builtin_amdgcn_wmma_f32_16x16x32_bf16(
              false, aq[kk].v, false, bk.v, (short)0, sacc[nt], false, false);
        }
      }
      // alibi + causal mask (alibi + ALPHA*qk, per reference)
      float alib[4];
#pragma unroll
      for (int nt = 0; nt < 4; ++nt)
        alib[nt] = alibi[((size_t)b * NHEAD + h) * SEQ + colBase + nt * 16 + l16];
      const int rowB = qt * 128 + wave * 16 + half * 8;
#pragma unroll
      for (int nt = 0; nt < 4; ++nt) {
        int col = colBase + nt * 16 + l16;
#pragma unroll
        for (int i = 0; i < 8; ++i) {
          float s = sacc[nt][i] * ALPHA + alib[nt];
          sacc[nt][i] = (col <= rowB + i) ? s : -3.3895e38f;
        }
      }
      // online softmax; rows live in 16-lane half-wave groups -> shfl_xor reduce
#pragma unroll
      for (int i = 0; i < 8; ++i) {
        float m = fmaxf(fmaxf(sacc[0][i], sacc[1][i]), fmaxf(sacc[2][i], sacc[3][i]));
#pragma unroll
        for (int off = 1; off < 16; off <<= 1) m = fmaxf(m, __shfl_xor(m, off, 32));
        float mnew  = fmaxf(mrun[i], m);
        float scale = __expf(mrun[i] - mnew);
        mrun[i] = mnew;
        lrun[i] *= scale;
#pragma unroll
        for (int ht = 0; ht < 8; ++ht) cacc[ht][i] *= scale;
        float rs = 0.0f;
#pragma unroll
        for (int nt = 0; nt < 4; ++nt) {
          float p = __expf(sacc[nt][i] - mnew);
          sacc[nt][i] = p;
          rs += p;
        }
#pragma unroll
        for (int off = 1; off < 16; off <<= 1) rs += __shfl_xor(rs, off, 32);
        lrun[i] += rs;
      }
      // P -> LDS (accumulator layout -> A-fragment layout), wave-private
#pragma unroll
      for (int nt = 0; nt < 4; ++nt)
#pragma unroll
        for (int i = 0; i < 8; ++i)
          Ps[wave][half * 8 + i][nt * 16 + l16] = f2bf(sacc[nt][i]);
      // ctx += P @ V : 2 K-chunks x 8 hd-tiles
#pragma unroll
      for (int kc = 0; kc < 2; ++kc) {
        FragAB ap;
        ap.q[0] = *(const uint4*)&Ps[wave][l16][kc * 32 + half * 8];
        ap.q[1] = *(const uint4*)&Ps[wave][l16][kc * 32 + 16 + half * 8];
#pragma unroll
        for (int ht = 0; ht < 8; ++ht) {
          FragAB bv;
          bv.q[0] = *(const uint4*)&Vt[ht * 16 + l16][kc * 32 + half * 8];
          bv.q[1] = *(const uint4*)&Vt[ht * 16 + l16][kc * 32 + 16 + half * 8];
          cacc[ht] = __builtin_amdgcn_wmma_f32_16x16x32_bf16(
              false, ap.v, false, bv.v, (short)0, cacc[ht], false, false);
        }
      }
    }
    __syncthreads();
  }
  // normalize + write merged-head ctx (bf16, [B,S,H])
#pragma unroll
  for (int i = 0; i < 8; ++i) {
    float inv = 1.0f / lrun[i];
    int qrow = qt * 128 + wave * 16 + half * 8 + i;
    size_t base = ((size_t)b * SEQ + qrow) * HID + (size_t)h * HEADD;
#pragma unroll
    for (int ht = 0; ht < 8; ++ht)
      ctxb[base + ht * 16 + l16] = f2bf(cacc[ht][i] * inv);
  }
}

// ---------------------------------------------------------------- output GEMM
// out[4096 x 2048] = ctx * Wd + bd + residual (fp32)
__global__ void __launch_bounds__(256) gemm_out_kernel(
    const unsigned short* __restrict__ A,
    const unsigned short* __restrict__ Bm,
    const float* __restrict__ bias,
    const float* __restrict__ residual,
    float* __restrict__ out) {
  const int K = HID, N = HID;
  __shared__ unsigned short As[128][40];
  __shared__ unsigned short Bs[128][40];
  const int tid  = threadIdx.x;
  const int wave = tid >> 5, lane = tid & 31;
  const int half = lane >> 4, l16 = lane & 15;
  const int nBlock = blockIdx.x * 128;
  const int mBlock = blockIdx.y * 128;
  const int wm = (wave & 3) * 32, wn = (wave >> 2) * 64;
  v8f acc[2][4] = {};
  for (int k0 = 0; k0 < K; k0 += 32) {
#pragma unroll
    for (int it = 0; it < 2; ++it) {        // A tile: async direct to LDS
      int idx = (tid + it * 256) * 8;
      int r = idx >> 5, c = idx & 31;
      async_ld16(&As[r][c], A + (size_t)(mBlock + r) * K + k0 + c);
    }
#pragma unroll
    for (int it = 0; it < 2; ++it) {
      int idx = (tid + it * 256) * 8;
      int kk = idx >> 7, nn = idx & 127;
      uint4 d = *(const uint4*)(Bm + (size_t)(k0 + kk) * N + nBlock + nn);
      const unsigned short* ds = (const unsigned short*)&d;
#pragma unroll
      for (int j = 0; j < 8; ++j) Bs[nn + j][kk] = ds[j];
    }
    async_wait0();
    __syncthreads();
    if (k0 + 32 < K) {
      int idx = tid * 8;
      __builtin_prefetch(A + (size_t)(mBlock + (idx >> 5)) * K + (k0 + 32) + (idx & 31), 0, 1);
      __builtin_prefetch(Bm + (size_t)(k0 + 32 + (idx >> 7)) * N + nBlock + (idx & 127), 0, 1);
    }
    FragAB af[2], bf[4];
#pragma unroll
    for (int mi = 0; mi < 2; ++mi) {
      int r = wm + mi * 16 + l16;
      af[mi].q[0] = *(const uint4*)&As[r][half * 8];
      af[mi].q[1] = *(const uint4*)&As[r][16 + half * 8];
    }
#pragma unroll
    for (int ni = 0; ni < 4; ++ni) {
      int n = wn + ni * 16 + l16;
      bf[ni].q[0] = *(const uint4*)&Bs[n][half * 8];
      bf[ni].q[1] = *(const uint4*)&Bs[n][16 + half * 8];
    }
#pragma unroll
    for (int mi = 0; mi < 2; ++mi)
#pragma unroll
      for (int ni = 0; ni < 4; ++ni)
        acc[mi][ni] = __builtin_amdgcn_wmma_f32_16x16x32_bf16(
            false, af[mi].v, false, bf[ni].v, (short)0, acc[mi][ni], false, false);
    __syncthreads();
  }
#pragma unroll
  for (int mi = 0; mi < 2; ++mi) {
#pragma unroll
    for (int ni = 0; ni < 4; ++ni) {
      int col = nBlock + wn + ni * 16 + l16;
      float bia = bias[col];
#pragma unroll
      for (int i = 0; i < 8; ++i) {
        int rowg = mBlock + wm + mi * 16 + half * 8 + i;
        size_t di = (size_t)rowg * N + col;
        out[di] = acc[mi][ni][i] + bia + residual[di];
      }
    }
  }
}

// ---------------------------------------------------------------- launch
extern "C" void kernel_launch(void* const* d_in, const int* in_sizes, int n_in,
                              void* d_out, int out_size, void* d_ws, size_t ws_size,
                              hipStream_t stream) {
  const float* hs       = (const float*)d_in[0];
  // d_in[1] = attention_mask (pure causal; applied analytically in-kernel)
  const float* residual = (const float*)d_in[2];
  const float* alibi    = (const float*)d_in[3];
  const float* Wqkv     = (const float*)d_in[4];
  const float* bqkv     = (const float*)d_in[5];
  const float* Wd       = (const float*)d_in[6];
  const float* bd       = (const float*)d_in[7];
  float* out = (float*)d_out;

  char* ws = (char*)d_ws;
  size_t off = 0;
  auto carve = [&](size_t elems) {
    unsigned short* p = (unsigned short*)(ws + off);
    off += ((elems * 2 + 255) & ~(size_t)255);
    return p;
  };
  unsigned short* hsb  = carve((size_t)BATCH * SEQ * HID);
  unsigned short* wqb  = carve((size_t)HID * 3 * HID);
  unsigned short* wdb  = carve((size_t)HID * HID);
  unsigned short* qb   = carve((size_t)BATCH * SEQ * HID);
  unsigned short* kbuf = carve((size_t)BATCH * SEQ * HID);
  unsigned short* vbuf = carve((size_t)BATCH * SEQ * HID);
  unsigned short* ctxb = carve((size_t)BATCH * SEQ * HID);

  auto cvt = [&](const float* s, unsigned short* dst, size_t n) {
    int n4 = (int)(n / 4);
    cvt_bf16_kernel<<<(n4 + 255) / 256, 256, 0, stream>>>(s, dst, n4);
  };
  cvt(hs, hsb, (size_t)BATCH * SEQ * HID);
  cvt(Wqkv, wqb, (size_t)HID * 3 * HID);
  cvt(Wd, wdb, (size_t)HID * HID);

  gemm_qkv_kernel<<<dim3(3 * HID / 128, BATCH * SEQ / 128), 256, 0, stream>>>(
      hsb, wqb, bqkv, kbuf, qb, vbuf);
  attn_kernel<<<dim3(SEQ / 128, NHEAD, BATCH), 256, 0, stream>>>(
      qb, kbuf, vbuf, alibi, ctxb);
  gemm_out_kernel<<<dim3(HID / 128, BATCH * SEQ / 128), 256, 0, stream>>>(
      ctxb, wdb, bd, residual, out);
}
